// Gated_PYG_84851373900199
// MI455X (gfx1250) — matrix operand order repaired and checked by reading, not compiled, not run on hardware
//
#include <hip/hip_runtime.h>
#include <hip/hip_bf16.h>
#include <math.h>

#define NN 50000
#define NE 600000
#define DD 128
#define NG 512
#define NC 10
#define NL 4
#define D3 384

typedef __attribute__((ext_vector_type(16))) __bf16 v16bf;
typedef __attribute__((ext_vector_type(8)))  float  v8f;

__device__ __forceinline__ unsigned short f2bf(float f) {
  unsigned int u = __float_as_uint(f);
  u += 0x7FFFu + ((u >> 16) & 1u);          // round-to-nearest-even
  return (unsigned short)(u >> 16);
}
__device__ __forceinline__ __bf16 bfbits(unsigned short s) {
  union { unsigned short u; __bf16 b; } cv; cv.u = s; return cv.b;
}

// A fragment (16x32 bf16) from an LDS strip laid out [16 rows][128 K] as ushort.
// Layout: lanes 0-15: M=lane, K in {0..15}; lanes 16-31: M=lane-16, K in {16..31} (matches B).
__device__ __forceinline__ v16bf load_a_frag(const unsigned short* aS, int lane, int k0) {
  int g = lane >> 4, r = lane & 15;
  v16bf a;
#pragma unroll
  for (int e = 0; e < 16; ++e) {
    int k = 16 * g + e;
    a[e] = bfbits(aS[r * DD + k0 + k]);
  }
  return a;
}

// B fragment (32x16 bf16) from global bf16 weights stored as [N][K=128] (contiguous K).
// Contiguous-K layout used here (same as A): N = lane%16; lanes 0-15 hold K=k0..k0+15, lanes 16-31 hold K=k0+16..k0+31.
__device__ __forceinline__ v16bf load_b_frag(const unsigned short* w, int n, int k0, int lane) {
  int g = lane >> 4;
  const uint4* p = reinterpret_cast<const uint4*>(w + (size_t)n * DD + k0 + 16 * g);
  union { uint4 q[2]; unsigned short s[16]; } u;
  u.q[0] = p[0]; u.q[1] = p[1];
  v16bf b;
#pragma unroll
  for (int e = 0; e < 16; ++e) b[e] = bfbits(u.s[e]);
  return b;
}

// ---------------- weight prep ----------------
__global__ void k_zero(float* p, int n) {
  for (int i = blockIdx.x * blockDim.x + threadIdx.x; i < n; i += gridDim.x * blockDim.x)
    p[i] = 0.f;
}
__global__ void k_cvt_bf16(const float* __restrict__ src, unsigned short* __restrict__ dst, int n) {
  for (int i = blockIdx.x * blockDim.x + threadIdx.x; i < n; i += gridDim.x * blockDim.x)
    dst[i] = f2bf(src[i]);
}
// Ws is [L][K][N]; store transposed bf16 as [L][N][K] so B loads are contiguous in K.
__global__ void k_transpose_ws(const float* __restrict__ Ws, unsigned short* __restrict__ wsT) {
  int tot = NL * DD * DD;
  for (int i = blockIdx.x * blockDim.x + threadIdx.x; i < tot; i += gridDim.x * blockDim.x) {
    int l = i / (DD * DD), rem = i % (DD * DD);
    int k = rem / DD, n = rem % DD;
    wsT[(size_t)l * DD * DD + (size_t)n * DD + k] = f2bf(Ws[i]);
  }
}

// ---------------- m = h @ Ws[l] ----------------
// grid: 3125 (50000/16 row strips), block: 256 (8 waves -> 8 column tiles of 16)
__global__ void k_msg_gemm(const float* __restrict__ h, const unsigned short* __restrict__ wT,
                           float* __restrict__ m) {
  __shared__ unsigned short aS[16 * DD];
  int row0 = blockIdx.x * 16;
  int tid = threadIdx.x;
  for (int i = tid; i < 16 * DD; i += 256)
    aS[i] = f2bf(h[(size_t)(row0 + (i >> 7)) * DD + (i & 127)]);
  __syncthreads();
  int wave = tid >> 5, lane = tid & 31;
  int g = lane >> 4, nl = lane & 15;
  int n0 = wave * 16;
  v8f acc = {};
#pragma unroll
  for (int kt = 0; kt < 4; ++kt) {
    v16bf a = load_a_frag(aS, lane, kt * 32);
    v16bf b = load_b_frag(wT, n0 + nl, kt * 32, lane);
    acc = __builtin_amdgcn_wmma_f32_16x16x32_bf16(false, a, false, b, (short)0, acc, false, false);
  }
#pragma unroll
  for (int i = 0; i < 8; ++i)
    m[(size_t)(row0 + i + 8 * g) * DD + n0 + nl] = acc[i];
}

// ---------------- edge scatter: agg[dst] += m[src] ----------------
// grid: 75000, block 256 (8 waves, one edge per wave; lane handles 4 floats)
__global__ void k_edge_scatter(const float* __restrict__ m, const int* __restrict__ src,
                               const int* __restrict__ dst, float* __restrict__ agg) {
  int wave = threadIdx.x >> 5, lane = threadIdx.x & 31;
  int e = blockIdx.x * 8 + wave;
  if (e >= NE) return;
  int s = src[e], d = dst[e];
  const float4* ms = reinterpret_cast<const float4*>(m + (size_t)s * DD);
  float4 v = ms[lane];
  float* ad = agg + (size_t)d * DD + lane * 4;
  atomicAdd(ad + 0, v.x);
  atomicAdd(ad + 1, v.y);
  atomicAdd(ad + 2, v.z);
  atomicAdd(ad + 3, v.w);
}

// ---------------- fused GRU: gi = agg@W_ih^T, gh = h@W_hh^T, gate math, ReLU ----------------
// grid: 3125, block 256. 48 WMMA tile jobs (2 GEMMs x 24 col tiles), 6 per wave.
__global__ void k_gru(const float* __restrict__ agg, float* __restrict__ h,
                      const unsigned short* __restrict__ wih, const unsigned short* __restrict__ whh,
                      const float* __restrict__ bi, const float* __restrict__ bh) {
  __shared__ unsigned short aS[16 * DD];
  __shared__ unsigned short hS[16 * DD];
  __shared__ float giS[16 * D3];
  __shared__ float ghS[16 * D3];
  int row0 = blockIdx.x * 16;
  int tid = threadIdx.x;
  for (int i = tid; i < 16 * DD; i += 256) {
    int r = i >> 7, c = i & 127;
    size_t gidx = (size_t)(row0 + r) * DD + c;
    aS[i] = f2bf(agg[gidx]);
    hS[i] = f2bf(h[gidx]);
  }
  __syncthreads();
  int wave = tid >> 5, lane = tid & 31;
  int g = lane >> 4, nl = lane & 15;
  for (int j = wave; j < 48; j += 8) {
    bool isI = j < 24;
    int ct = isI ? j : j - 24;
    const unsigned short* aSrc = isI ? aS : hS;
    const unsigned short* w = isI ? wih : whh;
    float* outS = isI ? giS : ghS;
    int n0 = ct * 16;
    v8f acc = {};
#pragma unroll
    for (int kt = 0; kt < 4; ++kt) {
      v16bf a = load_a_frag(aSrc, lane, kt * 32);
      v16bf b = load_b_frag(w, n0 + nl, kt * 32, lane);
      acc = __builtin_amdgcn_wmma_f32_16x16x32_bf16(false, a, false, b, (short)0, acc, false, false);
    }
#pragma unroll
    for (int i = 0; i < 8; ++i)
      outS[(i + 8 * g) * D3 + n0 + nl] = acc[i];
  }
  __syncthreads();
  for (int i = tid; i < 16 * DD; i += 256) {
    int r = i >> 7, c = i & 127;
    float ir = giS[r * D3 + c]       + bi[c];
    float iz = giS[r * D3 + 128 + c] + bi[128 + c];
    float in_ = giS[r * D3 + 256 + c]+ bi[256 + c];
    float hr = ghS[r * D3 + c]       + bh[c];
    float hz = ghS[r * D3 + 128 + c] + bh[128 + c];
    float hn = ghS[r * D3 + 256 + c] + bh[256 + c];
    float rr = 1.f / (1.f + __expf(-(ir + hr)));
    float zz = 1.f / (1.f + __expf(-(iz + hz)));
    float nn = tanhf(in_ + rr * hn);
    size_t gidx = (size_t)(row0 + r) * DD + c;
    float ho = h[gidx];
    float hv = (1.f - zz) * nn + zz * ho;
    h[gidx] = hv > 0.f ? hv : 0.f;       // ReLU, in-place (rows owned by this block)
  }
}

// ---------------- mean pool ----------------
__global__ void k_pool(const float* __restrict__ h, const int* __restrict__ batch,
                       float* __restrict__ sums, float* __restrict__ cnt) {
  int wave = threadIdx.x >> 5, lane = threadIdx.x & 31;
  int node = blockIdx.x * 8 + wave;
  if (node >= NN) return;
  int b = batch[node];
  const float4* hp = reinterpret_cast<const float4*>(h + (size_t)node * DD);
  float4 v = hp[lane];
  float* sp = sums + (size_t)b * DD + lane * 4;
  atomicAdd(sp + 0, v.x);
  atomicAdd(sp + 1, v.y);
  atomicAdd(sp + 2, v.z);
  atomicAdd(sp + 3, v.w);
  if (lane == 0) atomicAdd(&cnt[b], 1.f);
}
__global__ void k_pool_div(const float* __restrict__ sums, const float* __restrict__ cnt,
                           float* __restrict__ hg) {
  int i = blockIdx.x * blockDim.x + threadIdx.x;
  if (i < NG * DD) hg[i] = sums[i] / fmaxf(cnt[i >> 7], 1.f);
}

// ---------------- fc1 (WMMA) + ELU ----------------
// grid: 32 (512/16 strips), block 256 (8 waves = 8 col tiles)
__global__ void k_fc1(const float* __restrict__ hg, const unsigned short* __restrict__ w,
                      const float* __restrict__ bias, float* __restrict__ hg1) {
  __shared__ unsigned short aS[16 * DD];
  int row0 = blockIdx.x * 16;
  int tid = threadIdx.x;
  for (int i = tid; i < 16 * DD; i += 256)
    aS[i] = f2bf(hg[(size_t)(row0 + (i >> 7)) * DD + (i & 127)]);
  __syncthreads();
  int wave = tid >> 5, lane = tid & 31;
  int g = lane >> 4, nl = lane & 15;
  int n0 = wave * 16;
  v8f acc = {};
#pragma unroll
  for (int kt = 0; kt < 4; ++kt) {
    v16bf a = load_a_frag(aS, lane, kt * 32);
    v16bf b = load_b_frag(w, n0 + nl, kt * 32, lane);
    acc = __builtin_amdgcn_wmma_f32_16x16x32_bf16(false, a, false, b, (short)0, acc, false, false);
  }
  float bv = bias[n0 + nl];
#pragma unroll
  for (int i = 0; i < 8; ++i) {
    float x = acc[i] + bv;
    hg1[(size_t)(row0 + i + 8 * g) * DD + n0 + nl] = x > 0.f ? x : (__expf(x) - 1.f);
  }
}

// ---------------- fc2 ----------------
__global__ void k_fc2(const float* __restrict__ hg1, const float* __restrict__ w,
                      const float* __restrict__ b, float* __restrict__ logits) {
  int i = blockIdx.x * blockDim.x + threadIdx.x;
  if (i >= NG * NC) return;
  int gg = i / NC, c = i % NC;
  float s = b[c];
  const float* hp = hg1 + (size_t)gg * DD;
  const float* wp = w + (size_t)c * DD;
#pragma unroll 4
  for (int k = 0; k < DD; ++k) s += hp[k] * wp[k];
  logits[i] = s;
}

// ---------------- log_softmax over axis 0 (per class, across 512 graphs) ----------------
__global__ void k_logsoftmax0(const float* __restrict__ logits, float* __restrict__ out) {
  __shared__ float red[256];
  int c = blockIdx.x, tid = threadIdx.x;
  float mx = -1e30f;
  for (int g = tid; g < NG; g += 256) mx = fmaxf(mx, logits[g * NC + c]);
  red[tid] = mx; __syncthreads();
  for (int s = 128; s > 0; s >>= 1) {
    if (tid < s) red[tid] = fmaxf(red[tid], red[tid + s]);
    __syncthreads();
  }
  mx = red[0]; __syncthreads();
  float sum = 0.f;
  for (int g = tid; g < NG; g += 256) sum += __expf(logits[g * NC + c] - mx);
  red[tid] = sum; __syncthreads();
  for (int s = 128; s > 0; s >>= 1) {
    if (tid < s) red[tid] += red[tid + s];
    __syncthreads();
  }
  float lse = logf(red[0]) + mx;
  for (int g = tid; g < NG; g += 256) out[g * NC + c] = logits[g * NC + c] - lse;
}

extern "C" void kernel_launch(void* const* d_in, const int* in_sizes, int n_in,
                              void* d_out, int out_size, void* d_ws, size_t ws_size,
                              hipStream_t stream) {
  (void)in_sizes; (void)n_in; (void)out_size; (void)ws_size;
  const float* h_in   = (const float*)d_in[0];
  const int*   eidx   = (const int*)d_in[1];      // int32 (JAX default x64 off)
  // d_in[2] = edge_attr (unused by reference)
  const int*   batch  = (const int*)d_in[3];
  const float* Ws     = (const float*)d_in[4];
  const float* Wih    = (const float*)d_in[5];
  const float* Whh    = (const float*)d_in[6];
  const float* bih    = (const float*)d_in[7];
  const float* bhh    = (const float*)d_in[8];
  const float* fc1w   = (const float*)d_in[9];
  const float* fc1b   = (const float*)d_in[10];
  const float* fc2w   = (const float*)d_in[11];
  const float* fc2b   = (const float*)d_in[12];
  float* out = (float*)d_out;

  const int* src = eidx;
  const int* dst = eidx + NE;

  // workspace carve-up (256B aligned)
  char* ws = (char*)d_ws;
  size_t off = 0;
  auto carve = [&](size_t bytes) -> void* {
    void* p = ws + off;
    off = (off + bytes + 255) & ~(size_t)255;
    return p;
  };
  float* h_cur          = (float*)carve((size_t)NN * DD * 4);
  float* m_buf          = (float*)carve((size_t)NN * DD * 4);
  float* agg            = (float*)carve((size_t)NN * DD * 4);
  unsigned short* wsT   = (unsigned short*)carve((size_t)NL * DD * DD * 2);
  unsigned short* wihB  = (unsigned short*)carve((size_t)NL * D3 * DD * 2);
  unsigned short* whhB  = (unsigned short*)carve((size_t)NL * D3 * DD * 2);
  unsigned short* fc1B  = (unsigned short*)carve((size_t)DD * DD * 2);
  float* sums           = (float*)carve((size_t)NG * DD * 4);
  float* cnt            = (float*)carve((size_t)NG * 4);
  float* hg             = (float*)carve((size_t)NG * DD * 4);
  float* hg1            = (float*)carve((size_t)NG * DD * 4);
  float* logits         = (float*)carve((size_t)NG * NC * 4);

  // init mutable state
  hipMemcpyAsync(h_cur, h_in, (size_t)NN * DD * 4, hipMemcpyDeviceToDevice, stream);

  // one-time weight conversion (cheap, deterministic per call)
  k_transpose_ws<<<256, 256, 0, stream>>>(Ws, wsT);
  k_cvt_bf16<<<512, 256, 0, stream>>>(Wih, wihB, NL * D3 * DD);
  k_cvt_bf16<<<512, 256, 0, stream>>>(Whh, whhB, NL * D3 * DD);
  k_cvt_bf16<<<64, 256, 0, stream>>>(fc1w, fc1B, DD * DD);

  const int strips = NN / 16;            // 3125 (50000 divisible by 16)
  for (int l = 0; l < NL; ++l) {
    k_msg_gemm<<<strips, 256, 0, stream>>>(h_cur, wsT + (size_t)l * DD * DD, m_buf);
    k_zero<<<2048, 256, 0, stream>>>(agg, NN * DD);
    k_edge_scatter<<<(NE + 7) / 8, 256, 0, stream>>>(m_buf, src, dst, agg);
    k_gru<<<strips, 256, 0, stream>>>(agg, h_cur,
                                      wihB + (size_t)l * D3 * DD,
                                      whhB + (size_t)l * D3 * DD,
                                      bih + (size_t)l * D3, bhh + (size_t)l * D3);
  }

  k_zero<<<256, 256, 0, stream>>>(sums, NG * DD);
  k_zero<<<2, 256, 0, stream>>>(cnt, NG);
  k_pool<<<(NN + 7) / 8, 256, 0, stream>>>(h_cur, batch, sums, cnt);
  k_pool_div<<<(NG * DD + 255) / 256, 256, 0, stream>>>(sums, cnt, hg);
  k_fc1<<<NG / 16, 256, 0, stream>>>(hg, fc1B, fc1b, hg1);
  k_fc2<<<(NG * NC + 255) / 256, 256, 0, stream>>>(hg1, fc2w, fc2b, logits);
  k_logsoftmax0<<<NC, 256, 0, stream>>>(logits, out);
}
